// BaseWindowAttention_5669356834455
// MI455X (gfx1250) — matrix operand order repaired and checked by reading, not compiled, not run on hardware
//
#include <hip/hip_runtime.h>
#include <hip/hip_bf16.h>

typedef __attribute__((ext_vector_type(16))) _Float16 v16h;
typedef __attribute__((ext_vector_type(8)))  _Float16 v8h;
typedef __attribute__((ext_vector_type(8)))  float    v8f;
typedef __attribute__((ext_vector_type(4)))  unsigned int v4u;
typedef __attribute__((ext_vector_type(8)))  int v8i;
typedef __attribute__((ext_vector_type(4)))  int v4i;

__device__ __forceinline__ v8f wmma_f16(v16h a, v16h b, v8f c) {
  // D = A(16x32 f16) * B(32x16 f16) + C(16x16 f32)
  return __builtin_amdgcn_wmma_f32_16x16x32_f16(false, a, false, b, (short)0, c, false, false);
}

__device__ __forceinline__ v16h cat8(v8h a, v8h b) {
  return __builtin_shufflevector(a, b, 0,1,2,3,4,5,6,7,8,9,10,11,12,13,14,15);
}

__device__ __forceinline__ float half16_max(float v) {
#pragma unroll
  for (int d = 1; d < 16; d <<= 1) v = fmaxf(v, __shfl_xor(v, d, 32));
  return v;
}
__device__ __forceinline__ float half16_sum(float v) {
#pragma unroll
  for (int d = 1; d < 16; d <<= 1) v += __shfl_xor(v, d, 32);
  return v;
}

// ---- Tensor Data Mover: 2D tile load Global -> LDS (D# per CDNA5 ISA) ----
// Group0: [1:0]=count=1 | [63:32]=lds_addr | [120:64]=global_addr | [127:126]=type=2
// Group1: [17:16]=data_size(1=2B) | [79:48]=tensor_dim0 | [111:80]=tensor_dim1 |
//         [127:112]=tile_dim0 | [143:128]=tile_dim1 | [159:144]=tile_dim2(0) |
//         [207:160]=tensor_dim0_stride | rest 0
__device__ __forceinline__ void tdm_load_2d(unsigned lds_off, const void* gptr,
                                            unsigned tensor_d0, unsigned tensor_d1,
                                            unsigned tile_d0, unsigned tile_d1,
                                            unsigned long long stride0_elems) {
  unsigned long long ga = (unsigned long long)gptr;
  v4u g0;
  g0[0] = 1u;                                    // count=1, user descriptor
  g0[1] = lds_off;                               // LDS byte address (dynamic LDS base = 0)
  g0[2] = (unsigned)ga;
  g0[3] = (unsigned)((ga >> 32) & 0x01FFFFFFull) | (2u << 30);  // addr[56:32] | type=2
  v8i g1;
  g1[0] = (int)(1u << 16);                                        // data_size = 2 bytes
  g1[1] = (int)((tensor_d0 & 0xFFFFu) << 16);                     // tensor_dim0[15:0]
  g1[2] = (int)((tensor_d0 >> 16) | ((tensor_d1 & 0xFFFFu) << 16));
  g1[3] = (int)((tensor_d1 >> 16) | (tile_d0 << 16));             // tile_dim0
  g1[4] = (int)(tile_d1);                                         // tile_dim1, tile_dim2=0
  g1[5] = (int)(stride0_elems & 0xFFFFFFFFull);                   // tensor_dim0_stride
  g1[6] = (int)((stride0_elems >> 32) & 0xFFFFull);
  g1[7] = 0;
  v4i z4 = {0, 0, 0, 0};
  v8i z8 = {0, 0, 0, 0, 0, 0, 0, 0};
  __builtin_amdgcn_tensor_load_to_lds(g0, g1, z4, z4, z8, 0);
}

// ---------------- conversion kernels ----------------
__global__ __launch_bounds__(256) void cvt_f16_kernel(const float* __restrict__ in,
                                                      _Float16* __restrict__ out, int n) {
  int i = blockIdx.x * blockDim.x + threadIdx.x;
  if (i < n) out[i] = (_Float16)in[i];
}

// in: [K][N] fp32 row-major  ->  out: [N][K] f16 (transposed)
__global__ __launch_bounds__(256) void transpose_cvt_kernel(const float* __restrict__ in,
                                                            _Float16* __restrict__ out,
                                                            int K, int N) {
  int idx = blockIdx.x * blockDim.x + threadIdx.x;
  if (idx < K * N) {
    int n = idx / K, k = idx % K;
    out[(size_t)n * K + k] = (_Float16)in[(size_t)k * N + n];
  }
}

// ---------------- TDM-staged, double-buffered WMMA GEMM (K fixed at 256) -------
// C[M][N] = A[M][256] * Bt[N][256]^T (+ bias). Block: 8 waves, tile 128(M) x 64(N).
// Dynamic LDS (64KB): B panel [64][256]f16 @ 0 (one TDM at start);
//                     A tiles [2][128][64]f16 @ 32768 / 49152 (double-buffered).
// Pipeline: 4 K-panels of 64; stage i issues A(i+1) then s_wait_tensorcnt(1)
// (TDM completes in-order per wave, so everything except the new tile is done).
template<bool OUT_F16, bool BIAS>
__global__ __launch_bounds__(256)
void gemm_wmma_kernel(const _Float16* __restrict__ A, const _Float16* __restrict__ Bt,
                      void* __restrict__ Cv, const float* __restrict__ bias,
                      int M, int N) {
  constexpr int K = 256;
  extern __shared__ char smem[];

  const int lane = threadIdx.x & 31;
  const int wave = threadIdx.x >> 5;
  const int half = lane >> 4;
  const int l16  = lane & 15;
  const int mblk = blockIdx.x * 128;
  const int n0   = blockIdx.y * 64;

  v8f acc[4] = {v8f{}, v8f{}, v8f{}, v8f{}};

  if (wave == 0) {  // prologue: whole B panel + first A tile
    tdm_load_2d(0,     Bt + (size_t)n0 * K,   K, 64,  K,  64,  (unsigned long long)K);
    tdm_load_2d(32768, A  + (size_t)mblk * K, K, 128, 64, 128, (unsigned long long)K);
  }

#pragma unroll
  for (int kp = 0; kp < 4; ++kp) {          // K-panels of 64
    const int cur = kp & 1;
    if (wave == 0) {
      if (kp < 3) {  // stage next A tile into the other buffer
        tdm_load_2d(cur ? 32768u : 49152u,
                    A + (size_t)mblk * K + (kp + 1) * 64, K, 128, 64, 128,
                    (unsigned long long)K);
        __builtin_amdgcn_s_wait_tensorcnt((short)1);
      } else {
        __builtin_amdgcn_s_wait_tensorcnt((short)0);
      }
    }
    __syncthreads();  // staged data visible to all waves

    const _Float16* aBuf  = (const _Float16*)(smem + 32768 + cur * 16384) +
                            (size_t)(wave * 16 + l16) * 64;
    const _Float16* bBase = (const _Float16*)smem + kp * 64 + half * 16;
#pragma unroll
    for (int k32 = 0; k32 < 64; k32 += 32) {
      v16h af = *(const v16h*)(aBuf + k32 + half * 16);
#pragma unroll
      for (int nt = 0; nt < 4; ++nt) {
        const _Float16* b = bBase + (size_t)(nt * 16 + l16) * K + k32;
        acc[nt] = wmma_f16(af, *(const v16h*)b, acc[nt]);
      }
    }
    __syncthreads();  // all waves done reading before A buffer is overwritten
  }

  const int rbase = mblk + wave * 16 + half * 8;  // D layout rows
#pragma unroll
  for (int nt = 0; nt < 4; ++nt) {
    const int col = n0 + nt * 16 + l16;
    float bv = 0.f;
    if (BIAS) bv = bias[col];
#pragma unroll
    for (int r = 0; r < 8; ++r) {
      const size_t off = (size_t)(rbase + r) * N + col;
      if (OUT_F16) ((_Float16*)Cv)[off] = (_Float16)acc[nt][r];
      else         ((float*)Cv)[off]    = acc[nt][r] + bv;
    }
  }
}

// ---------------- window attention ----------------
// grid: (1024 windows, 2 head-groups); block: 128 threads = 4 waves = 4 heads.
// dynamic LDS: bias 4096*f32 (16K) | P 4*64*64*f16 (32K) | Vt 4*32*64*f16 (16K) = 64K
__global__ __launch_bounds__(128)
void window_attn_kernel(const _Float16* __restrict__ qkv,   // [65536][768] f16
                        const float* __restrict__ pos,      // [15][15]
                        _Float16* __restrict__ attn_out) {  // [65536][256] f16
  extern __shared__ char smem[];
  float*    sbias = (float*)smem;                       // 64*64 f32
  _Float16* sP    = (_Float16*)(smem + 16384);          // wave-local 64*64 f16
  _Float16* sVt   = (_Float16*)(smem + 16384 + 32768);  // wave-local 32*64 f16

  const int tid  = threadIdx.x;
  const int lane = tid & 31, wave = tid >> 5;
  const int half = lane >> 4, l16 = lane & 15;
  const int m = blockIdx.y * 4 + wave;                  // head

  const int wblk = blockIdx.x;
  const int img = wblk >> 6, wi = wblk & 63;
  const int wy = wi >> 3, wx = wi & 7;
  const size_t tokbase = (size_t)img * 4096 + (size_t)(wy * 8) * 64 + (size_t)(wx * 8);
  auto gtok = [&](int j) -> size_t { return tokbase + (size_t)((j >> 3) * 64 + (j & 7)); };

  // relative-position bias table, shared by all heads in block
  for (int e = tid; e < 4096; e += 128) {
    int i = e >> 6, j = e & 63;
    sbias[e] = pos[((i >> 3) - (j >> 3) + 7) * 15 + ((i & 7) - (j & 7) + 7)];
  }
  __syncthreads();

  const _Float16* qb = qkv + 0   + m * 32;
  const _Float16* kb = qkv + 256 + m * 32;
  const _Float16* vb = qkv + 512 + m * 32;

  // stage V transposed into LDS: Vt[c][j]
  _Float16* myVt = sVt + wave * 2048;
#pragma unroll
  for (int rr = 0; rr < 2; ++rr) {
    const int j = lane + rr * 32;
    const _Float16* vrow = vb + gtok(j) * 768;
#pragma unroll
    for (int cc = 0; cc < 4; ++cc) {
      v8h vv = *(const v8h*)(vrow + cc * 8);
#pragma unroll
      for (int e = 0; e < 8; ++e) myVt[(size_t)(cc * 8 + e) * 64 + j] = vv[e];
    }
  }

  // K fragments (B layout): lane half selects c-range 0-15 / 16-31
  v16h kf[4];
#pragma unroll
  for (int jt = 0; jt < 4; ++jt) {
    const int j = jt * 16 + l16;
    kf[jt] = *(const v16h*)(kb + gtok(j) * 768 + half * 16);
  }

  const float scale = 0.17677669529663687f;  // 32^-0.5
  _Float16* myP = sP + wave * 4096;

  // scores + softmax per 16-row tile (rows are independent)
  for (int it = 0; it < 4; ++it) {
    const int irow = it * 16 + l16;
    const _Float16* qrow = qb + gtok(irow) * 768;
    v16h qf = *(const v16h*)(qrow + half * 16);
    v8f s[4];
#pragma unroll
    for (int jt = 0; jt < 4; ++jt) { v8f z{}; s[jt] = wmma_f16(qf, kf[jt], z); }

    const int row_l = half * 8;
#pragma unroll
    for (int r = 0; r < 8; ++r) {
      const int i = it * 16 + row_l + r;
      float vals[4];
#pragma unroll
      for (int jt = 0; jt < 4; ++jt)
        vals[jt] = s[jt][r] * scale + sbias[i * 64 + jt * 16 + l16];
      float mx = fmaxf(fmaxf(vals[0], vals[1]), fmaxf(vals[2], vals[3]));
      mx = half16_max(mx);
      float es[4], sum = 0.f;
#pragma unroll
      for (int jt = 0; jt < 4; ++jt) { es[jt] = __expf(vals[jt] - mx); sum += es[jt]; }
      sum = half16_sum(sum);
      const float inv = 1.0f / sum;
#pragma unroll
      for (int jt = 0; jt < 4; ++jt)
        myP[(size_t)i * 64 + jt * 16 + l16] = (_Float16)(es[jt] * inv);
    }
  }
  asm volatile("s_wait_dscnt 0" ::: "memory");  // P/Vt relayout crosses lanes in-wave

  // V fragments from LDS (B layout, K = token dim)
  v16h vf[2][2];
#pragma unroll
  for (int kk = 0; kk < 2; ++kk)
#pragma unroll
    for (int nt = 0; nt < 2; ++nt) {
      const int c = nt * 16 + l16;
      vf[kk][nt] = *(const v16h*)(myVt + (size_t)c * 64 + kk * 32 + half * 16);
    }

  // O = P @ V
  for (int it = 0; it < 4; ++it) {
    v8f o0{}, o1{};
    const int prow = it * 16 + l16;
#pragma unroll
    for (int kk = 0; kk < 2; ++kk) {
      const _Float16* pr = myP + (size_t)prow * 64 + kk * 32;
      v16h pf = *(const v16h*)(pr + half * 16);
      o0 = wmma_f16(pf, vf[kk][0], o0);
      o1 = wmma_f16(pf, vf[kk][1], o1);
    }
    const int row_l = half * 8;
#pragma unroll
    for (int r = 0; r < 8; ++r) {
      const int i = it * 16 + row_l + r;
      const size_t t = gtok(i);
      attn_out[t * 256 + m * 32 + l16]      = (_Float16)o0[r];
      attn_out[t * 256 + m * 32 + 16 + l16] = (_Float16)o1[r];
    }
  }
}

// ---------------- launch ----------------
extern "C" void kernel_launch(void* const* d_in, const int* in_sizes, int n_in,
                              void* d_out, int out_size, void* d_ws, size_t ws_size,
                              hipStream_t stream) {
  const float* x    = (const float*)d_in[0];   // [2,8,64,64,256]
  const float* wqkv = (const float*)d_in[1];   // [256,768]
  const float* wout = (const float*)d_in[2];   // [256,256]
  const float* bout = (const float*)d_in[3];   // [256]
  const float* pos  = (const float*)d_in[4];   // [15,15]
  (void)in_sizes; (void)n_in; (void)out_size; (void)ws_size;

  char* ws = (char*)d_ws;
  _Float16* xh    = (_Float16*)(ws);                                            // 33,554,432 B
  _Float16* wqkvt = (_Float16*)(ws + 33554432);                                 //    393,216 B
  _Float16* woutt = (_Float16*)(ws + 33554432 + 393216);                        //    131,072 B
  _Float16* qkv   = (_Float16*)(ws + 33554432 + 393216 + 131072);               // 100,663,296 B
  _Float16* attn  = (_Float16*)(ws + 33554432 + 393216 + 131072 + 100663296);   // 33,554,432 B

  const int M = 65536;  // tokens

  cvt_f16_kernel<<<65536, 256, 0, stream>>>(x, xh, M * 256);
  transpose_cvt_kernel<<<768, 256, 0, stream>>>(wqkv, wqkvt, 256, 768);
  transpose_cvt_kernel<<<256, 256, 0, stream>>>(wout, woutt, 256, 256);

  // qkv = xh @ wqkv   (f16 out)  -- TDM-staged double-buffered GEMM
  gemm_wmma_kernel<true, false><<<dim3(512, 12), 256, 65536, stream>>>(
      xh, wqkvt, (void*)qkv, nullptr, M, 768);

  // windowed attention (f16 out)
  window_attn_kernel<<<dim3(1024, 2), 128, 65536, stream>>>(qkv, pos, attn);

  // out = attn @ wout + bout   (f32 out)
  gemm_wmma_kernel<false, true><<<dim3(512, 4), 256, 65536, stream>>>(
      attn, woutt, d_out, bout, M, 256);
}
